// SinkhornTransformerSeq2SeqDecoder_69020124446901
// MI455X (gfx1250) — matrix operand order repaired and checked by reading, not compiled, not run on hardware
//
#include <hip/hip_runtime.h>
#include <hip/hip_bf16.h>
#include <stdint.h>

// ---------------------------------------------------------------------------
// Types for CDNA5 WMMA (wave32): bf16 A/B fragments (16 x __bf16), f32x8 C/D.
// ---------------------------------------------------------------------------
typedef __bf16 bf16_t;
typedef __attribute__((ext_vector_type(16))) __bf16 bf16x16;
typedef __attribute__((ext_vector_type(8)))  float   f32x8;

union Frag {
  bf16x16 v;
  uint4   u[2];
};

#define NEG_INF_F (-1.0e9f)

__device__ __forceinline__ f32x8 wmma_bf16(const Frag& a, const Frag& b, f32x8 c) {
  // D = A(16x32 bf16) * B(32x16 bf16) + C(16x16 f32)
  return __builtin_amdgcn_wmma_f32_16x16x32_bf16(false, a.v, false, b.v,
                                                 (short)0, c, false, false);
}

// CDNA5 async global->LDS copy (16B per lane), tracked by ASYNCcnt.
__device__ __forceinline__ void async_load_b128(uint32_t lds_byte_off,
                                                const void* gaddr) {
  asm volatile("global_load_async_to_lds_b128 %0, %1, off"
               :: "v"(lds_byte_off), "v"(gaddr)
               : "memory");
}
__device__ __forceinline__ void wait_asynccnt0() {
  asm volatile("s_wait_asynccnt 0x0" ::: "memory");
}

__device__ __forceinline__ float gelu_tanh(float x) {
  float x3 = x * x * x;
  return 0.5f * x * (1.0f + tanhf(0.7978845608f * (x + 0.044715f * x3)));
}

// ---------------------------------------------------------------------------
// GEMM: C[2048,N] = act(scale*(A[2048,K] @ W[K,N]) + bias) (+res)
// A: bf16 row-major.  Wt: bf16 W^T, [N,K] row-major (packed by pack_wt_kernel).
// Fragment layouts (per CDNA5 ISA):
//   A frag (16x32): lane m=l&15, kh=l>>4; elems 0..7 -> K=kk+8*kh+0..7,
//                   elems 8..15 -> K=kk+16+8*kh+0..7  (two 16B runs)
//   B frag (32x16): lane n=l&15, kh=l>>4; elems 0..7 -> K=kk+8*kh+0..7, elems 8..15 -> K=kk+16+8*kh+0..7
//   C/D frag: row = v + 8*(l>>4), col = l&15
// Block: 256 thr = 8 waves -> 128 rows x 64 cols. The 64-col B strip is shared
// by all 8 waves, so it is staged once per K-step into LDS with async DMA
// (double-buffered: DMA of tile i+1 overlaps ds_load+WMMA on tile i).
// ---------------------------------------------------------------------------
__global__ void gemm_bf16_kernel(const bf16_t* __restrict__ A,
                                 const bf16_t* __restrict__ Wt,
                                 const float*  __restrict__ bias,
                                 const float*  __restrict__ res,
                                 float*        __restrict__ outf,
                                 bf16_t*       __restrict__ outb,
                                 int K, int N, float scale, int fuse) {
  __shared__ bf16_t Bs[2][32 * 64];  // two 4KB tiles: [n(64)][k(32)] bf16

  const int tid  = threadIdx.x;
  const int wid  = tid >> 5;
  const int lane = tid & 31;
  const int l15  = lane & 15;
  const int kh   = lane >> 4;
  const int m0   = blockIdx.y * 128 + wid * 16;
  const int n0   = blockIdx.x * 64;

  // Async-staging assignment: thread -> (row sn of B strip, 16B chunk sc).
  const int sn = tid >> 2;  // 0..63
  const int sc = tid & 3;   // 0..3
  const uint32_t lds0 = (uint32_t)(size_t)(&Bs[0][0]);
  const uint32_t lds_dst = lds0 + (uint32_t)(sn * 64 + sc * 16);  // bytes
  const bf16_t* gstage = Wt + (size_t)(n0 + sn) * K + sc * 8;

  // Prologue: kick off DMA of K-step 0 into buffer 0.
  async_load_b128(lds_dst, gstage);

  const bf16_t* arow = A + (size_t)(m0 + l15) * K;
  Frag a_cur;
  {
    const bf16_t* pa = arow + 8 * kh;
    a_cur.u[0] = *(const uint4*)(pa);
    a_cur.u[1] = *(const uint4*)(pa + 16);
  }

  f32x8 c[4] = {};
  const int nsteps = K >> 5;
  for (int i = 0; i < nsteps; ++i) {
    // Hand-off: my DMA for tile i done, then all waves' DMAs done.
    wait_asynccnt0();
    __syncthreads();
    // Overlap: start DMA of tile i+1 into the other buffer.
    if (i + 1 < nsteps)
      async_load_b128(lds0 + (uint32_t)(((i + 1) & 1) * 4096 + sn * 64 + sc * 16),
                      gstage + (i + 1) * 32);
    // Prefetch next A fragment (registers) while computing on a_cur.
    Frag a_next = a_cur;
    if (i + 1 < nsteps) {
      const bf16_t* pa = arow + (i + 1) * 32 + 8 * kh;
      a_next.u[0] = *(const uint4*)(pa);
      a_next.u[1] = *(const uint4*)(pa + 16);
    }
    const bf16_t* bbuf = &Bs[i & 1][0];
#pragma unroll
    for (int nt = 0; nt < 4; ++nt) {
      Frag b;
      const bf16_t* pb = bbuf + (nt * 16 + l15) * 32 + 8 * kh;
      b.u[0] = *(const uint4*)(pb);       // ds_load_b128
      b.u[1] = *(const uint4*)(pb + 16);  // ds_load_b128
      c[nt] = wmma_bf16(a_cur, b, c[nt]);
    }
    a_cur = a_next;
  }
#pragma unroll
  for (int nt = 0; nt < 4; ++nt) {
    const int col = n0 + nt * 16 + l15;
#pragma unroll
    for (int v = 0; v < 8; ++v) {
      const int row = m0 + v + 8 * kh;
      float val = c[nt][v] * scale;
      if (bias) val += bias[col];
      if (fuse == 1) val = gelu_tanh(val);
      const size_t o = (size_t)row * N + col;
      if (res)  val += res[o];
      if (outf) outf[o] = val;
      if (outb) outb[o] = (bf16_t)val;
    }
  }
}

// Pack W[K,N] f32 -> Wt[N,K] bf16 (transpose + convert).
__global__ void pack_wt_kernel(const float* __restrict__ W,
                               bf16_t* __restrict__ Wt, int K, int N) {
  int idx = blockIdx.x * blockDim.x + threadIdx.x;
  if (idx >= K * N) return;
  int n = idx / K, k = idx - n * K;
  Wt[idx] = (bf16_t)W[(size_t)k * N + n];
}

// ---------------------------------------------------------------------------
// Fused attention (flash-style online softmax), one block per (b, h, qtile64).
// mode 0: sinkhorn self-attn, 2 source tiles (local causal+mask, sorted n>0).
// mode 1: cross-attn, 16 source tiles streamed over S_SRC=1024 with pad mask.
// K buffers: bf16 [b, s, h*64+dh] row-major (B-frag = contiguous dh run).
// VT buffers: bf16 [b, h, dh, s] (B-frag for PV = contiguous s run).
// ---------------------------------------------------------------------------
__global__ void attn_kernel(const bf16_t* __restrict__ Q,
                            const bf16_t* __restrict__ K0,
                            const bf16_t* __restrict__ K1,
                            const bf16_t* __restrict__ VT0,
                            const bf16_t* __restrict__ VT1,
                            const float*  __restrict__ maskv,
                            bf16_t*       __restrict__ out,
                            int mode) {
  const int qt = blockIdx.x, h = blockIdx.y, b = blockIdx.z;
  const int tid = threadIdx.x;
  const int wid = tid >> 5;
  const int lane = tid & 31;
  const int l15 = lane & 15;
  const int kh  = lane >> 4;
  const int mt  = wid & 3;   // row tile (16 rows) of the 64-row query block
  const int ng  = wid >> 2;  // column-half group

  __shared__ bf16_t Pb[64 * 64];
  __shared__ float rowmax[64], rowsum[64], rowscale[64], rmnew[64];
  __shared__ float wred[8][16];

  if (tid < 64) { rowmax[tid] = -3.0e38f; rowsum[tid] = 0.0f; }

  // Q fragments held in registers across the whole source loop (Dh=64 => 2 ksteps)
  Frag qf[2];
  {
    const bf16_t* qrow = Q + ((size_t)(b * 1024 + qt * 64 + mt * 16 + l15)) * 512 + h * 64;
#pragma unroll
    for (int ks = 0; ks < 2; ++ks) {
      const bf16_t* p = qrow + ks * 32 + 8 * kh;
      qf[ks].u[0] = *(const uint4*)(p);
      qf[ks].u[1] = *(const uint4*)(p + 16);
    }
  }
  f32x8 o[2] = {};
  const int ntiles = (mode == 0) ? 2 : 16;
  __syncthreads();

  for (int j = 0; j < ntiles; ++j) {
    const bf16_t* kbase;
    if (mode == 0)
      kbase = ((j == 0) ? K0 : K1) + ((size_t)(b * 1024 + qt * 64)) * 512 + h * 64;
    else
      kbase = K0 + ((size_t)(b * 1024 + j * 64)) * 512 + h * 64;

    // S = Q K^T for this wave's two 16x16 score tiles
    f32x8 s2[2] = {};
#pragma unroll
    for (int it = 0; it < 2; ++it) {
      const int tcol = (2 * ng + it) * 16 + l15;
      const bf16_t* kp = kbase + (size_t)tcol * 512;
#pragma unroll
      for (int ks = 0; ks < 2; ++ks) {
        Frag kbfr;
        const bf16_t* p = kp + ks * 32 + 8 * kh;
        kbfr.u[0] = *(const uint4*)(p);
        kbfr.u[1] = *(const uint4*)(p + 16);
        s2[it] = wmma_bf16(qf[ks], kbfr, s2[it]);
      }
    }
    // masking (finite -1e9 to reproduce reference softmax semantics)
#pragma unroll
    for (int it = 0; it < 2; ++it) {
      const int tcol = (2 * ng + it) * 16 + l15;
      float mval = 1.0f;
      if (mode == 0) { if (j == 0) mval = maskv[b * 1024 + qt * 64 + tcol]; }
      else           { mval = maskv[b * 1024 + j * 64 + tcol]; }
#pragma unroll
      for (int v = 0; v < 8; ++v) {
        const int rloc = mt * 16 + v + 8 * kh;
        bool ok;
        if (mode == 0) ok = (j == 0) ? ((tcol <= rloc) && (mval > 0.0f)) : (qt > 0);
        else           ok = (mval > 0.0f);
        if (!ok) s2[it][v] = NEG_INF_F;
      }
    }
    // per-row tile max (shuffle within 16-lane halves, then cross-wave via LDS)
    float pr[8];
#pragma unroll
    for (int v = 0; v < 8; ++v) pr[v] = fmaxf(s2[0][v], s2[1][v]);
#pragma unroll
    for (int off = 1; off < 16; off <<= 1)
#pragma unroll
      for (int v = 0; v < 8; ++v) pr[v] = fmaxf(pr[v], __shfl_xor(pr[v], off, 32));
    if (l15 == 0)
#pragma unroll
      for (int v = 0; v < 8; ++v) wred[wid][kh * 8 + v] = pr[v];
    __syncthreads();
    if (tid < 64) {
      const int rmt = tid >> 4, ri = tid & 15;
      float tm = fmaxf(wred[rmt][ri], wred[rmt + 4][ri]);
      float nm = fmaxf(rowmax[tid], tm);
      rmnew[tid] = nm;
      float sc = __expf(rowmax[tid] - nm);
      rowscale[tid] = sc;
      rowmax[tid] = nm;
      rowsum[tid] *= sc;
    }
    __syncthreads();
    // P = exp(S - m_new): write bf16 P tile to LDS, accumulate row sums
#pragma unroll
    for (int it = 0; it < 2; ++it) {
      const int tcol = (2 * ng + it) * 16 + l15;
#pragma unroll
      for (int v = 0; v < 8; ++v) {
        const int rloc = mt * 16 + v + 8 * kh;
        float pv = __expf(s2[it][v] - rmnew[rloc]);
        s2[it][v] = pv;
        Pb[rloc * 64 + tcol] = (bf16_t)pv;
      }
    }
#pragma unroll
    for (int v = 0; v < 8; ++v) pr[v] = s2[0][v] + s2[1][v];
#pragma unroll
    for (int off = 1; off < 16; off <<= 1)
#pragma unroll
      for (int v = 0; v < 8; ++v) pr[v] += __shfl_xor(pr[v], off, 32);
    if (l15 == 0)
#pragma unroll
      for (int v = 0; v < 8; ++v) wred[wid][kh * 8 + v] = pr[v];
    // rescale running O accumulators
#pragma unroll
    for (int dt = 0; dt < 2; ++dt)
#pragma unroll
      for (int v = 0; v < 8; ++v) o[dt][v] *= rowscale[mt * 16 + v + 8 * kh];
    __syncthreads();
    if (tid < 64) rowsum[tid] += wred[tid >> 4][tid & 15] + wred[(tid >> 4) + 4][tid & 15];

    // O += P V  (A-frags from LDS P, B-frags from transposed V)
    const bf16_t* vbase;
    if (mode == 0)
      vbase = ((j == 0) ? VT0 : VT1) + ((size_t)((b * 8 + h) * 64)) * 1024 + qt * 64;
    else
      vbase = VT0 + ((size_t)((b * 8 + h) * 64)) * 1024 + j * 64;
#pragma unroll
    for (int dt = 0; dt < 2; ++dt) {
      const int dh = (2 * ng + dt) * 16 + l15;
      const bf16_t* vp = vbase + (size_t)dh * 1024;
#pragma unroll
      for (int ks = 0; ks < 2; ++ks) {
        Frag pf;
        const bf16_t* pp = &Pb[(mt * 16 + l15) * 64 + ks * 32 + 8 * kh];
        pf.u[0] = *(const uint4*)(pp);
        pf.u[1] = *(const uint4*)(pp + 16);
        Frag vf;
        const bf16_t* vq = vp + ks * 32 + 8 * kh;
        vf.u[0] = *(const uint4*)(vq);
        vf.u[1] = *(const uint4*)(vq + 16);
        o[dt] = wmma_bf16(pf, vf, o[dt]);
      }
    }
    __syncthreads();
  }
  // normalize and emit bf16 (input of the W_O GEMM)
#pragma unroll
  for (int dt = 0; dt < 2; ++dt) {
    const int dh = (2 * ng + dt) * 16 + l15;
#pragma unroll
    for (int v = 0; v < 8; ++v) {
      const int rloc = mt * 16 + v + 8 * kh;
      float val = o[dt][v] / rowsum[rloc];
      out[((size_t)(b * 1024 + qt * 64 + rloc)) * 512 + h * 64 + dh] = (bf16_t)val;
    }
  }
}

// ---------------------------------------------------------------------------
// LayerNorm over EMB=512: one block per row; emits f32 (optional) + bf16.
// ---------------------------------------------------------------------------
__global__ void layernorm_kernel(const float* __restrict__ x,
                                 const float* __restrict__ scale,
                                 const float* __restrict__ bias,
                                 float* __restrict__ outf,
                                 bf16_t* __restrict__ outb) {
  const int row = blockIdx.x;
  const int t = threadIdx.x;  // 256
  __shared__ float red[256];
  __shared__ float stat[2];
  const float* xr = x + (size_t)row * 512;
  float a = xr[t], b2 = xr[t + 256];
  red[t] = a + b2;
  __syncthreads();
  for (int off = 128; off > 0; off >>= 1) {
    if (t < off) red[t] += red[t + off];
    __syncthreads();
  }
  if (t == 0) stat[0] = red[0] * (1.0f / 512.0f);
  __syncthreads();
  float mu = stat[0];
  float da = a - mu, db = b2 - mu;
  red[t] = da * da + db * db;
  __syncthreads();
  for (int off = 128; off > 0; off >>= 1) {
    if (t < off) red[t] += red[t + off];
    __syncthreads();
  }
  if (t == 0) stat[1] = rsqrtf(red[0] * (1.0f / 512.0f) + 1e-6f);
  __syncthreads();
  float rs = stat[1];
  float o1 = da * rs * scale[t] + bias[t];
  float o2 = db * rs * scale[t + 256] + bias[t + 256];
  if (outf) { outf[(size_t)row * 512 + t] = o1; outf[(size_t)row * 512 + t + 256] = o2; }
  outb[(size_t)row * 512 + t] = (bf16_t)o1;
  outb[(size_t)row * 512 + t + 256] = (bf16_t)o2;
}

// ---------------------------------------------------------------------------
// Small supporting kernels.
// ---------------------------------------------------------------------------
__global__ void embed_kernel(const float* __restrict__ emb,
                             const float* __restrict__ pos,
                             const int* __restrict__ targets,
                             float* __restrict__ y) {
  int idx = blockIdx.x * blockDim.x + threadIdx.x;  // 2048*512
  int d = idx & 511;
  int s = (idx >> 9) & 1023;
  int b = idx >> 19;
  int yid = (s == 0) ? 0 : targets[b * 1024 + s - 1];  // shift_right
  y[idx] = emb[(size_t)yid * 512 + d] + pos[s * 512 + d];
}

__global__ void mask_kernel(const int* __restrict__ t, float* __restrict__ m) {
  int i = blockIdx.x * blockDim.x + threadIdx.x;  // 2048
  m[i] = (t[i] > 0) ? 1.0f : 0.0f;
}

__global__ void f32_to_bf16_kernel(const float* __restrict__ in,
                                   bf16_t* __restrict__ out, int n) {
  int i = blockIdx.x * blockDim.x + threadIdx.x;
  if (i < n) out[i] = (bf16_t)in[i];
}

// bf16 [B,1024,512] -> bf16 [B,8,64,1024]  (per-head transpose for V)
__global__ void transpose_heads_kernel(const bf16_t* __restrict__ in,
                                       bf16_t* __restrict__ out) {
  int idx = blockIdx.x * blockDim.x + threadIdx.x;  // 1,048,576
  int s  = idx & 1023;
  int dh = (idx >> 10) & 63;
  int h  = (idx >> 16) & 7;
  int b  = idx >> 19;
  out[idx] = in[((size_t)(b * 1024 + s)) * 512 + h * 64 + dh];
}

// ksum[b,n,h,dh] = sum_t k[b, n*64+t, h, dh] * mask[b, n*64+t]
__global__ void ksum_kernel(const float* __restrict__ k,
                            const float* __restrict__ mask,
                            float* __restrict__ ks) {
  int idx = blockIdx.x * blockDim.x + threadIdx.x;  // 16384
  int dh = idx & 63;
  int h  = (idx >> 6) & 7;
  int n  = (idx >> 9) & 15;
  int b  = idx >> 13;
  float s = 0.0f;
  for (int t = 0; t < 64; ++t) {
    int srow = b * 1024 + n * 64 + t;
    s += k[(size_t)srow * 512 + h * 64 + dh] * mask[srow];
  }
  ks[idx] = s;
}

// logits[b,h,n,m] = sum_k ksum[b,n,h,k]*sort_w[k,m] + sort_b[m]; m<n else -1e9
__global__ void sortlogits_kernel(const float* __restrict__ ks,
                                  const float* __restrict__ sw,
                                  const float* __restrict__ sb,
                                  float* __restrict__ lg) {
  int idx = blockIdx.x * blockDim.x + threadIdx.x;  // 4096
  int m = idx & 15;
  int n = (idx >> 4) & 15;
  int h = (idx >> 8) & 7;
  int b = idx >> 11;
  float acc = sb[m];
  const float* kp = ks + (((size_t)(b * 16 + n)) * 8 + h) * 64;
  for (int kk = 0; kk < 64; ++kk) acc += kp[kk] * sw[kk * 16 + m];
  if (!(m < n)) acc = NEG_INF_F;
  lg[idx] = acc;
}

// 5 Sinkhorn iterations on 16x16 log-alpha, one block per (b,h).
__global__ void sinkhorn_kernel(const float* __restrict__ lg, float* __restrict__ perm) {
  int bh = blockIdx.x;
  int t = threadIdx.x;  // 256
  __shared__ float la[256];
  __shared__ float red[16];
  la[t] = lg[bh * 256 + t];
  __syncthreads();
  for (int it = 0; it < 5; ++it) {
    if (t < 16) {  // row logsumexp (axis -1)
      float m = -3.0e38f;
      for (int j = 0; j < 16; ++j) m = fmaxf(m, la[t * 16 + j]);
      float s = 0.0f;
      for (int j = 0; j < 16; ++j) s += __expf(la[t * 16 + j] - m);
      red[t] = m + __logf(s);
    }
    __syncthreads();
    la[t] -= red[t >> 4];
    __syncthreads();
    if (t < 16) {  // column logsumexp (axis -2)
      float m = -3.0e38f;
      for (int i = 0; i < 16; ++i) m = fmaxf(m, la[i * 16 + t]);
      float s = 0.0f;
      for (int i = 0; i < 16; ++i) s += __expf(la[i * 16 + t] - m);
      red[t] = m + __logf(s);
    }
    __syncthreads();
    la[t] -= red[t & 15];
    __syncthreads();
  }
  perm[bh * 256 + t] = __expf(la[t]);
}

// sk/sv: apply soft permutation to masked K/V blocks.
// sk -> bf16 key layout [b, n*64+t, h*64+dh]; sv -> bf16 V^T layout [b,h,dh,s].
__global__ void permapply_kernel(const float* __restrict__ perm,
                                 const float* __restrict__ k,
                                 const float* __restrict__ v,
                                 const float* __restrict__ mask,
                                 bf16_t* __restrict__ skb,
                                 bf16_t* __restrict__ svt) {
  int idx = blockIdx.x * blockDim.x + threadIdx.x;  // 1,048,576
  int d = idx & 511;
  int t = (idx >> 9) & 63;
  int n = (idx >> 15) & 15;
  int b = idx >> 19;
  int h = d >> 6, dh = d & 63;
  const float* pr = perm + (((size_t)(b * 8 + h)) * 16 + n) * 16;
  float ka = 0.0f, va = 0.0f;
  for (int m = 0; m < 16; ++m) {
    int srow = b * 1024 + m * 64 + t;
    float w = pr[m] * mask[srow];
    ka += w * k[(size_t)srow * 512 + d];
    va += w * v[(size_t)srow * 512 + d];
  }
  skb[((size_t)(b * 1024 + n * 64 + t)) * 512 + d] = (bf16_t)ka;
  svt[(((size_t)(b * 8 + h)) * 64 + dh) * 1024 + n * 64 + t] = (bf16_t)va;
}

// ---------------------------------------------------------------------------
// Host orchestration.
// Input pytree flatten order (jax canonical: sorted dict keys, list order):
//  0 embed, 1 final_ln_b, 2 final_ln_s,
//  per layer L (base = 3 + 20L):
//   +0 cross.wk +1 cross.wo +2 cross.wq +3 cross.wv
//   +4 ln1_b +5 ln1_s +6 ln2_b +7 ln2_s +8 ln3_b +9 ln3_s
//   +10 mlp.b1 +11 mlp.b2 +12 mlp.w1 +13 mlp.w2
//   +14 sa.sort_b +15 sa.sort_w +16 sa.wk +17 sa.wo +18 sa.wq +19 sa.wv
//  123 logit_b, 124 logit_w, 125 pos_emb, 126 encoded, 127 src_mask, 128 targets
// ---------------------------------------------------------------------------
extern "C" void kernel_launch(void* const* d_in, const int* in_sizes, int n_in,
                              void* d_out, int out_size, void* d_ws, size_t ws_size,
                              hipStream_t stream) {
  (void)in_sizes; (void)n_in; (void)out_size; (void)ws_size;
  auto F = [&](int i) { return (const float*)d_in[i]; };

  const float* p_embed   = F(0);
  const float* p_flnb    = F(1);
  const float* p_flns    = F(2);
  const float* p_logit_b = F(123);
  const float* p_logit_w = F(124);
  const float* p_pos     = F(125);
  const float* p_enc     = F(126);
  const float* p_srcmask = F(127);
  const int*   p_targets = (const int*)d_in[128];

  char* wp_ = (char*)d_ws;
  auto alloc = [&](size_t bytes) -> void* {
    void* r = (void*)wp_;
    wp_ += (bytes + 255) & ~(size_t)255;
    return r;
  };
  const size_t MSD = 2048ull * 512;
  float*  yb  = (float*)alloc(MSD * 4);
  float*  xb  = (float*)alloc(MSD * 4);
  float*  zb  = (float*)alloc(MSD * 4);
  float*  kf  = (float*)alloc(MSD * 4);
  float*  vf  = (float*)alloc(MSD * 4);
  bf16_t* lnb = (bf16_t*)alloc(MSD * 2);
  bf16_t* qb  = (bf16_t*)alloc(MSD * 2);
  bf16_t* kb  = (bf16_t*)alloc(MSD * 2);
  bf16_t* vbr = (bf16_t*)alloc(MSD * 2);
  bf16_t* vtb = (bf16_t*)alloc(MSD * 2);
  bf16_t* skb = (bf16_t*)alloc(MSD * 2);
  bf16_t* svt = (bf16_t*)alloc(MSD * 2);
  bf16_t* ob  = (bf16_t*)alloc(MSD * 2);
  bf16_t* kcb = (bf16_t*)alloc(MSD * 2);
  bf16_t* vcb = (bf16_t*)alloc(MSD * 2);
  bf16_t* vct = (bf16_t*)alloc(MSD * 2);
  bf16_t* encb= (bf16_t*)alloc(MSD * 2);
  bf16_t* hb  = (bf16_t*)alloc(2048ull * 2048 * 2);
  bf16_t* wpk = (bf16_t*)alloc(512ull * 8000 * 2);
  float*  ksum= (float*)alloc(16384 * 4);
  float*  slog= (float*)alloc(4096 * 4);
  float*  perm= (float*)alloc(4096 * 4);
  float*  mbuf= (float*)alloc(2048 * 4);

  auto pack = [&](const float* W, int K, int N) {
    int tot = K * N;
    pack_wt_kernel<<<(tot + 255) / 256, 256, 0, stream>>>(W, wpk, K, N);
  };
  auto gemm = [&](const bf16_t* A, const float* bias, const float* res,
                  float* outf, bf16_t* outb, int K, int N, float scale, int fuse) {
    dim3 g(N / 64, 16);
    gemm_bf16_kernel<<<g, 256, 0, stream>>>(A, wpk, bias, res, outf, outb, K, N, scale, fuse);
  };

  // Prologue: mask, embedding+positional, encoder -> bf16.
  mask_kernel<<<8, 256, 0, stream>>>(p_targets, mbuf);
  embed_kernel<<<(int)(MSD / 256), 256, 0, stream>>>(p_embed, p_pos, p_targets, yb);
  f32_to_bf16_kernel<<<(int)(MSD / 256), 256, 0, stream>>>(p_enc, encb, (int)MSD);

  for (int L = 0; L < 6; ++L) {
    const int base = 3 + L * 20;
    const float* ca_wk = F(base + 0);
    const float* ca_wo = F(base + 1);
    const float* ca_wq = F(base + 2);
    const float* ca_wv = F(base + 3);
    const float* ln1b  = F(base + 4);
    const float* ln1s  = F(base + 5);
    const float* ln2b  = F(base + 6);
    const float* ln2s  = F(base + 7);
    const float* ln3b  = F(base + 8);
    const float* ln3s  = F(base + 9);
    const float* mb1   = F(base + 10);
    const float* mb2   = F(base + 11);
    const float* mw1   = F(base + 12);
    const float* mw2   = F(base + 13);
    const float* sortb = F(base + 14);
    const float* sortw = F(base + 15);
    const float* sa_wk = F(base + 16);
    const float* sa_wo = F(base + 17);
    const float* sa_wq = F(base + 18);
    const float* sa_wv = F(base + 19);

    // ---- Sinkhorn self-attention ----
    layernorm_kernel<<<2048, 256, 0, stream>>>(yb, ln1s, ln1b, nullptr, lnb);
    pack(sa_wq, 512, 512); gemm(lnb, nullptr, nullptr, nullptr, qb, 512, 512, 0.125f, 0);
    pack(sa_wk, 512, 512); gemm(lnb, nullptr, nullptr, kf, kb, 512, 512, 1.0f, 0);
    pack(sa_wv, 512, 512); gemm(lnb, nullptr, nullptr, vf, vbr, 512, 512, 1.0f, 0);
    transpose_heads_kernel<<<4096, 256, 0, stream>>>(vbr, vtb);
    ksum_kernel<<<64, 256, 0, stream>>>(kf, mbuf, ksum);
    sortlogits_kernel<<<16, 256, 0, stream>>>(ksum, sortw, sortb, slog);
    sinkhorn_kernel<<<16, 256, 0, stream>>>(slog, perm);
    permapply_kernel<<<4096, 256, 0, stream>>>(perm, kf, vf, mbuf, skb, svt);
    attn_kernel<<<dim3(16, 8, 2), 256, 0, stream>>>(qb, kb, skb, vtb, svt, mbuf, ob, 0);
    pack(sa_wo, 512, 512); gemm(ob, nullptr, yb, xb, nullptr, 512, 512, 1.0f, 0);  // +residual y

    // ---- Cross attention ----
    layernorm_kernel<<<2048, 256, 0, stream>>>(xb, ln2s, ln2b, nullptr, lnb);
    pack(ca_wq, 512, 512); gemm(lnb, nullptr, nullptr, nullptr, qb, 512, 512, 0.125f, 0);
    pack(ca_wk, 512, 512); gemm(encb, nullptr, nullptr, nullptr, kcb, 512, 512, 1.0f, 0);
    pack(ca_wv, 512, 512); gemm(encb, nullptr, nullptr, nullptr, vcb, 512, 512, 1.0f, 0);
    transpose_heads_kernel<<<4096, 256, 0, stream>>>(vcb, vct);
    attn_kernel<<<dim3(16, 8, 2), 256, 0, stream>>>(qb, kcb, nullptr, vct, nullptr, p_srcmask, ob, 1);
    pack(ca_wo, 512, 512); gemm(ob, nullptr, xb, zb, nullptr, 512, 512, 1.0f, 0);  // +residual x

    // ---- MLP ----
    layernorm_kernel<<<2048, 256, 0, stream>>>(zb, ln3s, ln3b, nullptr, lnb);
    pack(mw1, 512, 2048); gemm(lnb, mb1, nullptr, nullptr, hb, 512, 2048, 1.0f, 1);  // GELU
    pack(mw2, 2048, 512); gemm(hb, mb2, zb, yb, nullptr, 2048, 512, 1.0f, 0);        // +residual z
  }

  // Final LN + logits (N=8000, 125 column blocks).
  layernorm_kernel<<<2048, 256, 0, stream>>>(yb, p_flns, p_flnb, nullptr, lnb);
  pack(p_logit_w, 512, 8000);
  gemm(lnb, p_logit_b, nullptr, (float*)d_out, nullptr, 512, 8000, 1.0f, 0);
}
